// USAQueryDecoderBlock_44135083933750
// MI455X (gfx1250) — matrix operand order repaired and checked by reading, not compiled, not run on hardware
//
#include <hip/hip_runtime.h>
#include <hip/hip_bf16.h>
#include <math.h>

// ---------------------------------------------------------------------------
// Problem constants (match reference)
// ---------------------------------------------------------------------------
#define Bsz 2
#define NQ  512
#define NCx 2048
#define Dm  768
#define Hh  8
#define HDm 96
#define FFNm 3072

typedef __attribute__((ext_vector_type(16))) _Float16 v16h;
typedef __attribute__((ext_vector_type(8)))  float    v8f;

union F16Frag {
  v16h v;
  unsigned u[8];
  _Float16 h[16];
};

union V8H {
  int4 i4;
  _Float16 h[8];
};

union U32H {
  unsigned u;
  _Float16 h[2];
};

__device__ __forceinline__ v8f wmma_f16(v16h a, v16h b, v8f c) {
  // D = A(16x32 f16) * B(32x16 f16) + C(16x16 f32)
  return __builtin_amdgcn_wmma_f32_16x16x32_f16(false, a, false, b, (short)0, c,
                                                false, false);
}

__device__ __forceinline__ float gelu_exact(float x) {
  return 0.5f * x * (1.0f + erff(x * 0.70710678118654752f));
}

// A-fragment (16x32 f16) pair base K index for pair p (elements 2p,2p+1).
// Contiguous order: lanes 0-15 hold K {0..15}; lanes 16-31 hold {16..31}.
__device__ __forceinline__ int afrag_k0(int p, int hw) {
  return (hw << 4) + (p << 1);
}

// ---------------------------------------------------------------------------
// LayerNorm: f32 rows -> f16 rows
// ---------------------------------------------------------------------------
__global__ __launch_bounds__(256)
void ln_kernel(const float* __restrict__ x, const float* __restrict__ g,
               const float* __restrict__ bta, _Float16* __restrict__ y, int D) {
  __shared__ float red[256];
  const int row = blockIdx.x;
  const int tid = threadIdx.x;
  const float* xr = x + (size_t)row * D;

  float s = 0.f;
  for (int i = tid; i < D; i += 256) s += xr[i];
  red[tid] = s;
  __syncthreads();
  for (int o = 128; o > 0; o >>= 1) {
    if (tid < o) red[tid] += red[tid + o];
    __syncthreads();
  }
  const float mean = red[0] * (1.0f / (float)D);
  __syncthreads();

  float s2 = 0.f;
  for (int i = tid; i < D; i += 256) {
    float d = xr[i] - mean;
    s2 += d * d;
  }
  red[tid] = s2;
  __syncthreads();
  for (int o = 128; o > 0; o >>= 1) {
    if (tid < o) red[tid] += red[tid + o];
    __syncthreads();
  }
  const float var = red[0] * (1.0f / (float)D);
  const float inv = rsqrtf(var + 1e-5f);

  _Float16* yr = y + (size_t)row * D;
  for (int i = tid; i < D; i += 256)
    yr[i] = (_Float16)((xr[i] - mean) * inv * g[i] + bta[i]);
}

// ---------------------------------------------------------------------------
// Generic WMMA GEMM:  C[M,N] = act(A[M,K] * W[K,N] + bias) (+ residual)
//   A: f16 or f32 (aHalf flag), W/bias/residual: f32. Out f32 and/or f16.
//   Block = 128 threads (4 waves), tile 64(M) x 64(N), K-step 32.
// ---------------------------------------------------------------------------
__global__ __launch_bounds__(128)
void gemm_kernel(const void* __restrict__ Aptr, int aHalf,
                 const float* __restrict__ W, const float* __restrict__ bias,
                 const float* __restrict__ residual,
                 float* __restrict__ outF, _Float16* __restrict__ outH,
                 int M, int N, int K, int doGelu) {
  __shared__ __align__(16) _Float16 Atile[64][32];
  __shared__ __align__(16) _Float16 Wt[64][32];   // transposed: Wt[n][k]

  const int tid  = threadIdx.x;
  const int wave = tid >> 5;
  const int lane = tid & 31;
  const int hw   = lane >> 4;   // half-wave
  const int l16  = lane & 15;
  const int tM = blockIdx.y * 64;
  const int tN = blockIdx.x * 64;

  const _Float16* Ah = (const _Float16*)Aptr;
  const float*    Af = (const float*)Aptr;

  v8f acc[4];
#pragma unroll
  for (int nt = 0; nt < 4; ++nt)
#pragma unroll
    for (int r = 0; r < 8; ++r) acc[nt][r] = 0.f;

  for (int kc = 0; kc < K; kc += 32) {
    __syncthreads();
    {  // stage A (64 rows x 32 k) as f16
      const int row = tid >> 1, koff = (tid & 1) * 16;
      const size_t gbase = (size_t)(tM + row) * K + kc + koff;
      if (aHalf) {
#pragma unroll
        for (int i = 0; i < 16; ++i) Atile[row][koff + i] = Ah[gbase + i];
      } else {
#pragma unroll
        for (int i = 0; i < 16; ++i) Atile[row][koff + i] = (_Float16)Af[gbase + i];
      }
      if (kc + 32 < K) {  // CDNA5 global_prefetch of next K-chunk
        const void* pf = aHalf ? (const void*)(Ah + gbase + 32)
                               : (const void*)(Af + gbase + 32);
        __builtin_prefetch(pf, 0, 1);
      }
    }
    {  // stage W transposed (32 k x 64 n) -> Wt[n][k], f32 -> f16
      const int k = tid >> 2, n0 = (tid & 3) * 16;
      const size_t gbase = (size_t)(kc + k) * N + tN + n0;
#pragma unroll
      for (int i = 0; i < 16; ++i) Wt[n0 + i][k] = (_Float16)W[gbase + i];
    }
    __syncthreads();

    F16Frag a;
#pragma unroll
    for (int p = 0; p < 8; ++p)
      a.u[p] = *(const unsigned*)&Atile[wave * 16 + l16][afrag_k0(p, hw)];

#pragma unroll
    for (int nt = 0; nt < 4; ++nt) {
      F16Frag bw;
#pragma unroll
      for (int p = 0; p < 8; ++p)
        bw.u[p] = *(const unsigned*)&Wt[nt * 16 + l16][(hw << 4) + 2 * p];
      acc[nt] = wmma_f16(a.v, bw.v, acc[nt]);
    }
  }

  // epilogue
#pragma unroll
  for (int nt = 0; nt < 4; ++nt) {
#pragma unroll
    for (int r = 0; r < 8; ++r) {
      const int row = tM + wave * 16 + (hw << 3) + r;
      const int col = tN + nt * 16 + l16;
      float v = acc[nt][r] + bias[col];
      if (doGelu) v = gelu_exact(v);
      if (residual) v += residual[(size_t)row * N + col];
      if (outF) outF[(size_t)row * N + col] = v;
      if (outH) outH[(size_t)row * N + col] = (_Float16)v;
    }
  }
}

// ---------------------------------------------------------------------------
// Relative coordinate bias: gelu(delta @ w1 + b1) @ w2 + b2
//   -> bias16[b][h][q][c]  (f16)
// ---------------------------------------------------------------------------
__global__ __launch_bounds__(256)
void relbias_kernel(const float* __restrict__ qc, const float* __restrict__ cc,
                    const float* __restrict__ w1, const float* __restrict__ b1,
                    const float* __restrict__ w2, const float* __restrict__ b2,
                    _Float16* __restrict__ bias) {
  __shared__ float sw1[128], sb1[64], sw2[512], sb2[8];
  const int tid = threadIdx.x;
  for (int i = tid; i < 128; i += 256) sw1[i] = w1[i];
  for (int i = tid; i < 64; i += 256) sb1[i] = b1[i];
  for (int i = tid; i < 512; i += 256) sw2[i] = w2[i];
  if (tid < 8) sb2[tid] = b2[tid];
  __syncthreads();

  const size_t idx = (size_t)blockIdx.x * 256 + tid;  // Bsz*NQ*NCx total
  const int c = (int)(idx % NCx);
  const size_t t = idx / NCx;
  const int q = (int)(t % NQ);
  const int b = (int)(t / NQ);

  const float dx = qc[((size_t)b * NQ + q) * 2 + 0] - cc[((size_t)b * NCx + c) * 2 + 0];
  const float dy = qc[((size_t)b * NQ + q) * 2 + 1] - cc[((size_t)b * NCx + c) * 2 + 1];

  float accv[8];
#pragma unroll
  for (int hh = 0; hh < 8; ++hh) accv[hh] = sb2[hh];
  for (int j = 0; j < 64; ++j) {
    const float hv = gelu_exact(dx * sw1[j] + dy * sw1[64 + j] + sb1[j]);
#pragma unroll
    for (int hh = 0; hh < 8; ++hh) accv[hh] += hv * sw2[j * 8 + hh];
  }
#pragma unroll
  for (int hh = 0; hh < 8; ++hh)
    bias[(((size_t)(b * Hh + hh) * NQ) + q) * NCx + c] = (_Float16)accv[hh];
}

// ---------------------------------------------------------------------------
// Flash attention: 4 waves/block share one (batch, head); each wave owns a
// 16-query tile. K and V 32-key chunks are staged cooperatively into LDS with
// b128 global loads (V stored transposed so B-fragments are aligned b32 LDS
// reads). Online softmax uses 16-lane shuffle reductions matching the C/D
// register layout. Optional additive bias (cross-attention).
//   Q/K/V: (B,NCtx,D) f16, head h at column h*96. Out: (B,NQ,D) f16.
//   Grid: 128 blocks x 128 threads = Bsz*Hh*(NQ/16) wave-jobs exactly.
// ---------------------------------------------------------------------------
__global__ __launch_bounds__(128)
void flash_kernel(const _Float16* __restrict__ Q, const _Float16* __restrict__ Km,
                  const _Float16* __restrict__ Vm, const _Float16* __restrict__ Bias,
                  _Float16* __restrict__ Out, int NCtx) {
  __shared__ __align__(16) _Float16 Ks[32][96];      // [key][hd]
  __shared__ __align__(16) _Float16 Vt[96][32];      // [hd][key] (transposed)
  __shared__ __align__(16) _Float16 pTile[4][16][32];

  const int tid  = threadIdx.x;
  const int wave = tid >> 5;
  const int lane = tid & 31;
  const int hw   = lane >> 4;
  const int l16  = lane & 15;

  // Block-level (b, h); waves differ only in q-tile (consecutive jobs).
  const int jb = blockIdx.x * 4;
  const int hh = (jb >> 5) & 7;
  const int b  = jb >> 8;
  const int qbase = ((jb + wave) & 31) * 16;

  // --- load Q fragments (3 k-steps of 32 over HD=96), pre-scaled by HD^-1/2 ---
  const float scale = 0.1020620726159658f;  // 1/sqrt(96)
  F16Frag aq[3];
  const _Float16* qrow = Q + ((size_t)(b * NQ + qbase + l16)) * Dm + hh * HDm;
#pragma unroll
  for (int ks = 0; ks < 3; ++ks) {
#pragma unroll
    for (int p = 0; p < 8; ++p) {
      U32H qv;
      qv.u = *(const unsigned*)(qrow + ks * 32 + afrag_k0(p, hw));
      aq[ks].h[2 * p]     = (_Float16)((float)qv.h[0] * scale);
      aq[ks].h[2 * p + 1] = (_Float16)((float)qv.h[1] * scale);
    }
  }

  float mrow[8], lrow[8];
  v8f acc[6];
#pragma unroll
  for (int r = 0; r < 8; ++r) { mrow[r] = -1e30f; lrow[r] = 0.f; }
#pragma unroll
  for (int nt = 0; nt < 6; ++nt)
#pragma unroll
    for (int r = 0; r < 8; ++r) acc[nt][r] = 0.f;

  for (int cc = 0; cc < NCtx; cc += 32) {
    // ---- cooperative staging of K (32x96) and V^T (96x32) chunks ----
    __syncthreads();
    {
      const size_t kvbase = ((size_t)b * NCtx + cc) * Dm + hh * HDm;
#pragma unroll
      for (int i = 0; i < 3; ++i) {
        const int c   = tid + 128 * i;       // 0..383 chunks of 8 halves
        const int key = c / 12;
        const int off = (c % 12) * 8;
        V8H kv, vv;
        kv.i4 = *(const int4*)(Km + kvbase + (size_t)key * Dm + off);
        vv.i4 = *(const int4*)(Vm + kvbase + (size_t)key * Dm + off);
        *(int4*)&Ks[key][off] = kv.i4;
#pragma unroll
        for (int j = 0; j < 8; ++j) Vt[off + j][key] = vv.h[j];
        if (cc + 32 < NCtx) {
          __builtin_prefetch((const void*)(Km + kvbase + (size_t)(key + 32) * Dm + off), 0, 1);
          __builtin_prefetch((const void*)(Vm + kvbase + (size_t)(key + 32) * Dm + off), 0, 1);
        }
      }
    }
    __syncthreads();

    // ---- S = Q K^T for keys [cc, cc+32) : two 16x16 tiles ----
    v8f s0, s1;
#pragma unroll
    for (int r = 0; r < 8; ++r) { s0[r] = 0.f; s1[r] = 0.f; }
#pragma unroll
    for (int ks = 0; ks < 3; ++ks) {
      F16Frag bk0, bk1;
#pragma unroll
      for (int p = 0; p < 8; ++p) {
        const int kk = ks * 32 + (hw << 4) + 2 * p;
        bk0.u[p] = *(const unsigned*)&Ks[l16][kk];
        bk1.u[p] = *(const unsigned*)&Ks[16 + l16][kk];
      }
      s0 = wmma_f16(aq[ks].v, bk0.v, s0);
      s1 = wmma_f16(aq[ks].v, bk1.v, s1);
    }

    // ---- add bias (cross-attention only) ----
    if (Bias) {
      const _Float16* bp =
          Bias + (((size_t)(b * Hh + hh) * NQ) + qbase) * NCtx + cc;
#pragma unroll
      for (int r = 0; r < 8; ++r) {
        const size_t ro = (size_t)((hw << 3) + r) * NCtx;
        s0[r] += (float)bp[ro + l16];
        s1[r] += (float)bp[ro + 16 + l16];
      }
    }

    // ---- online softmax over this 32-key chunk ----
    float p0[8], p1[8], alpha[8];
#pragma unroll
    for (int r = 0; r < 8; ++r) {
      float v = fmaxf(s0[r], s1[r]);
      v = fmaxf(v, __shfl_xor(v, 1, 32));
      v = fmaxf(v, __shfl_xor(v, 2, 32));
      v = fmaxf(v, __shfl_xor(v, 4, 32));
      v = fmaxf(v, __shfl_xor(v, 8, 32));
      const float mnew = fmaxf(mrow[r], v);
      alpha[r] = __expf(mrow[r] - mnew);
      p0[r] = __expf(s0[r] - mnew);
      p1[r] = __expf(s1[r] - mnew);
      float rs = p0[r] + p1[r];
      rs += __shfl_xor(rs, 1, 32);
      rs += __shfl_xor(rs, 2, 32);
      rs += __shfl_xor(rs, 4, 32);
      rs += __shfl_xor(rs, 8, 32);
      lrow[r] = lrow[r] * alpha[r] + rs;
      mrow[r] = mnew;
    }
#pragma unroll
    for (int nt = 0; nt < 6; ++nt)
#pragma unroll
      for (int r = 0; r < 8; ++r) acc[nt][r] *= alpha[r];

    // ---- transpose probs through LDS into A-fragment layout ----
#pragma unroll
    for (int r = 0; r < 8; ++r) {
      const int row = (hw << 3) + r;
      pTile[wave][row][l16]      = (_Float16)p0[r];
      pTile[wave][row][l16 + 16] = (_Float16)p1[r];
    }
    __syncthreads();
    F16Frag pa;
#pragma unroll
    for (int p = 0; p < 8; ++p)
      pa.u[p] = *(const unsigned*)&pTile[wave][l16][afrag_k0(p, hw)];

    // ---- out += P * V (6 hd-tiles of 16) from transposed LDS V ----
#pragma unroll
    for (int nt = 0; nt < 6; ++nt) {
      F16Frag bv;
#pragma unroll
      for (int p = 0; p < 8; ++p)
        bv.u[p] = *(const unsigned*)&Vt[nt * 16 + l16][(hw << 4) + 2 * p];
      acc[nt] = wmma_f16(pa.v, bv.v, acc[nt]);
    }
  }

  // ---- normalize and write merged (B,NQ,D) f16 ----
  float inv[8];
#pragma unroll
  for (int r = 0; r < 8; ++r) inv[r] = 1.0f / lrow[r];
#pragma unroll
  for (int nt = 0; nt < 6; ++nt)
#pragma unroll
    for (int r = 0; r < 8; ++r) {
      const size_t row = (size_t)(b * NQ + qbase + (hw << 3) + r);
      Out[row * Dm + hh * HDm + nt * 16 + l16] = (_Float16)(acc[nt][r] * inv[r]);
    }
}

// ---------------------------------------------------------------------------
// Host-side orchestration
// ---------------------------------------------------------------------------
extern "C" void kernel_launch(void* const* d_in, const int* in_sizes, int n_in,
                              void* d_out, int out_size, void* d_ws, size_t ws_size,
                              hipStream_t stream) {
  (void)in_sizes; (void)n_in; (void)out_size; (void)ws_size;

  const float* query_state    = (const float*)d_in[0];
  const float* context_k      = (const float*)d_in[1];
  const float* context_v      = (const float*)d_in[2];
  const float* query_coords   = (const float*)d_in[3];
  const float* context_coords = (const float*)d_in[4];
  const float* sn_g = (const float*)d_in[5];
  const float* sn_b = (const float*)d_in[6];
  const float* sa_wq = (const float*)d_in[7];  const float* sa_bq = (const float*)d_in[8];
  const float* sa_wk = (const float*)d_in[9];  const float* sa_bk = (const float*)d_in[10];
  const float* sa_wv = (const float*)d_in[11]; const float* sa_bv = (const float*)d_in[12];
  const float* sa_wo = (const float*)d_in[13]; const float* sa_bo = (const float*)d_in[14];
  const float* cn_g = (const float*)d_in[15];
  const float* cn_b = (const float*)d_in[16];
  const float* ca_wq = (const float*)d_in[17]; const float* ca_bq = (const float*)d_in[18];
  const float* ca_wk = (const float*)d_in[19]; const float* ca_bk = (const float*)d_in[20];
  const float* ca_wv = (const float*)d_in[21]; const float* ca_bv = (const float*)d_in[22];
  const float* ca_wo = (const float*)d_in[23]; const float* ca_bo = (const float*)d_in[24];
  const float* rb_w1 = (const float*)d_in[25]; const float* rb_b1 = (const float*)d_in[26];
  const float* rb_w2 = (const float*)d_in[27]; const float* rb_b2 = (const float*)d_in[28];
  const float* fn_g = (const float*)d_in[29];
  const float* fn_b = (const float*)d_in[30];
  const float* ffn_w1 = (const float*)d_in[31]; const float* ffn_b1 = (const float*)d_in[32];
  const float* ffn_w2 = (const float*)d_in[33]; const float* ffn_b2 = (const float*)d_in[34];

  // workspace bump allocator (256B aligned)
  char* wp = (char*)d_ws;
  auto alloc = [&](size_t bytes) -> void* {
    void* r = (void*)wp;
    wp += (bytes + 255) & ~(size_t)255;
    return r;
  };
  _Float16* qn16   = (_Float16*)alloc((size_t)Bsz * NQ  * Dm   * 2);  // LN out (reused 3x)
  _Float16* q16    = (_Float16*)alloc((size_t)Bsz * NQ  * Dm   * 2);  // Q proj (self+cross)
  _Float16* kbuf   = (_Float16*)alloc((size_t)Bsz * NCx * Dm   * 2);  // K proj (self uses first part)
  _Float16* vbuf   = (_Float16*)alloc((size_t)Bsz * NCx * Dm   * 2);  // V proj
  _Float16* attn16 = (_Float16*)alloc((size_t)Bsz * NQ  * Dm   * 2);  // merged attn out
  _Float16* ffn1   = (_Float16*)alloc((size_t)Bsz * NQ  * FFNm * 2);  // gelu(hn@w1+b1)
  _Float16* bias16 = (_Float16*)alloc((size_t)Bsz * Hh * NQ * NCx * 2);
  float*    xbuf   = (float*)alloc((size_t)Bsz * NQ * Dm * 4);        // x after self-attn
  float*    x2buf  = (float*)alloc((size_t)Bsz * NQ * Dm * 4);        // x after cross-attn

  const int MQ = Bsz * NQ;   // 1024
  const int MC = Bsz * NCx;  // 4096
  const dim3 gB(128);
  const dim3 gemmQ(Dm / 64, MQ / 64);    // (12,16)
  const dim3 gemmC(Dm / 64, MC / 64);    // (12,64)
  const dim3 gemmF1(FFNm / 64, MQ / 64); // (48,16)

  // ---- self-attention ----
  ln_kernel<<<MQ, 256, 0, stream>>>(query_state, sn_g, sn_b, qn16, Dm);
  gemm_kernel<<<gemmQ, gB, 0, stream>>>(qn16, 1, sa_wq, sa_bq, nullptr, nullptr, q16,  MQ, Dm, Dm, 0);
  gemm_kernel<<<gemmQ, gB, 0, stream>>>(qn16, 1, sa_wk, sa_bk, nullptr, nullptr, kbuf, MQ, Dm, Dm, 0);
  gemm_kernel<<<gemmQ, gB, 0, stream>>>(qn16, 1, sa_wv, sa_bv, nullptr, nullptr, vbuf, MQ, Dm, Dm, 0);
  flash_kernel<<<128, gB, 0, stream>>>(q16, kbuf, vbuf, nullptr, attn16, NQ);
  gemm_kernel<<<gemmQ, gB, 0, stream>>>(attn16, 1, sa_wo, sa_bo, query_state, xbuf, nullptr, MQ, Dm, Dm, 0);

  // ---- relative coordinate bias ----
  relbias_kernel<<<(Bsz * NQ * NCx) / 256, 256, 0, stream>>>(
      query_coords, context_coords, rb_w1, rb_b1, rb_w2, rb_b2, bias16);

  // ---- cross-attention ----
  ln_kernel<<<MQ, 256, 0, stream>>>(xbuf, cn_g, cn_b, qn16, Dm);
  gemm_kernel<<<gemmQ, gB, 0, stream>>>(qn16, 1, ca_wq, ca_bq, nullptr, nullptr, q16,  MQ, Dm, Dm, 0);
  gemm_kernel<<<gemmC, gB, 0, stream>>>(context_k, 0, ca_wk, ca_bk, nullptr, nullptr, kbuf, MC, Dm, Dm, 0);
  gemm_kernel<<<gemmC, gB, 0, stream>>>(context_v, 0, ca_wv, ca_bv, nullptr, nullptr, vbuf, MC, Dm, Dm, 0);
  flash_kernel<<<128, gB, 0, stream>>>(q16, kbuf, vbuf, bias16, attn16, NCx);
  gemm_kernel<<<gemmQ, gB, 0, stream>>>(attn16, 1, ca_wo, ca_bo, xbuf, x2buf, nullptr, MQ, Dm, Dm, 0);

  // ---- FFN ----
  ln_kernel<<<MQ, 256, 0, stream>>>(x2buf, fn_g, fn_b, qn16, Dm);
  gemm_kernel<<<gemmF1, gB, 0, stream>>>(qn16, 1, ffn_w1, ffn_b1, nullptr, nullptr, ffn1, MQ, FFNm, Dm, 1);
  gemm_kernel<<<gemmQ, gB, 0, stream>>>(ffn1, 1, ffn_w2, ffn_b2, x2buf, (float*)d_out, nullptr, MQ, Dm, FFNm, 0);
}
